// Cellpred_27496380629540
// MI455X (gfx1250) — compile-verified, matrix operand order checked by reading, not run on hardware
//
#include <hip/hip_runtime.h>
#include <math.h>

// Problem constants (match reference)
#define Bc    512
#define Fc    1024
#define HISTc 20
#define FUTc  30
#define Lc    3
#define G4F   4096   // 4*F

typedef __attribute__((ext_vector_type(16))) __bf16 v16bf;
typedef __attribute__((ext_vector_type(8)))  __bf16 v8bf;
typedef __attribute__((ext_vector_type(8)))  float  v8f;

// ---------------- helpers ----------------

__device__ __forceinline__ unsigned short f2bf(float x) {
    union { float f; unsigned int u; } cv; cv.f = x;
    unsigned int u = cv.u;
    unsigned int r = u + 0x7fffu + ((u >> 16) & 1u);   // round-to-nearest-even
    return (unsigned short)(r >> 16);
}

// Load one WMMA bf16 fragment for this lane.
// Per the CDNA5 16-bit 16x32 layout, a lane holds 8 consecutive bf16 at K=kb
// and 8 consecutive at K=kb+16 (kb already includes the lane-half offset).
__device__ __forceinline__ v16bf load_frag(const __bf16* __restrict__ p) {
    v8bf lo = *(const v8bf*)p;          // global_load_b128
    v8bf hi = *(const v8bf*)(p + 16);   // global_load_b128
    v16bf r;
#pragma unroll
    for (int i = 0; i < 8; ++i) { r[i] = lo[i]; r[i + 8] = hi[i]; }
    return r;
}

// ---------------- conversion / init kernels ----------------

__global__ void k_f32_to_bf16(const float* __restrict__ src,
                              unsigned short* __restrict__ dst, int n) {
    int i = blockIdx.x * blockDim.x + threadIdx.x;
    int stride = gridDim.x * blockDim.x;
    for (; i < n; i += stride) dst[i] = f2bf(src[i]);
}

// seq_input is [B, F, HIST]; extract timestep t into contiguous bf16 [B*F]
__global__ void k_slice_x(const float* __restrict__ seq,
                          unsigned short* __restrict__ dst, int t) {
    int i = blockIdx.x * blockDim.x + threadIdx.x;
    if (i < Bc * Fc) dst[i] = f2bf(seq[(size_t)i * HISTc + t]);
}

__global__ void k_zero_state(float* __restrict__ h, float* __restrict__ c,
                             unsigned short* __restrict__ hbf) {
    int i = blockIdx.x * blockDim.x + threadIdx.x;
    if (i < Lc * Bc * Fc) { h[i] = 0.f; c[i] = 0.f; hbf[i] = 0; }
}

// ---------------- fused gates GEMM ----------------
// gates[M=512, N=4096] = X[512,1024] @ Wih^T + H[512,1024] @ Whh^T
// Wih/Whh are [N, K] row-major (== column-major K x N, exactly the B layout
// WMMA wants), so A and B fragments use the identical load pattern.
// Block: 256 threads = 8 waves arranged 2(M) x 4(N); block tile 128 x 128.
// Wave tile 64(M) x 32(N) = 4x2 WMMA tiles -> 6 fragment loads per 8 WMMAs,
// and each weight element is read only (512/64)=8x per layer-step from L2.

__global__ __launch_bounds__(256)
void k_gates_gemm(const unsigned short* __restrict__ Xu,
                  const unsigned short* __restrict__ Wihu,
                  const unsigned short* __restrict__ Hu,
                  const unsigned short* __restrict__ Whhu,
                  float* __restrict__ gates) {
    const __bf16* X   = (const __bf16*)Xu;
    const __bf16* Wih = (const __bf16*)Wihu;
    const __bf16* H   = (const __bf16*)Hu;
    const __bf16* Whh = (const __bf16*)Whhu;

    const int lane = threadIdx.x & 31;
    const int wave = threadIdx.x >> 5;
    const int wm   = wave >> 2;           // 0..1
    const int wn   = wave & 3;            // 0..3
    const int mBase = blockIdx.y * 128 + wm * 64;   // 4 M-tiles: +0,16,32,48
    const int nBase = blockIdx.x * 128 + wn * 32;   // 2 N-tiles: +0,16
    const int row   = lane & 15;
    const int kh    = (lane >> 4) << 3;   // 0 or 8: lane-half K offset

    v8f acc[4][2];
#pragma unroll
    for (int mt = 0; mt < 4; ++mt)
#pragma unroll
        for (int nt = 0; nt < 2; ++nt) acc[mt][nt] = (v8f){};

    // ---- X @ Wih^T ----
    {
        const __bf16* pa = X   + (size_t)(mBase + row) * Fc + kh;
        const __bf16* pb = Wih + (size_t)(nBase + row) * Fc + kh;
#pragma unroll 4
        for (int k = 0; k < Fc; k += 32) {
            v16bf fa[4], fb[2];
#pragma unroll
            for (int mt = 0; mt < 4; ++mt) fa[mt] = load_frag(pa + (size_t)(mt * 16) * Fc + k);
#pragma unroll
            for (int nt = 0; nt < 2; ++nt) fb[nt] = load_frag(pb + (size_t)(nt * 16) * Fc + k);
#pragma unroll
            for (int mt = 0; mt < 4; ++mt)
#pragma unroll
                for (int nt = 0; nt < 2; ++nt)
                    acc[mt][nt] = __builtin_amdgcn_wmma_f32_16x16x32_bf16(
                        false, fa[mt], false, fb[nt], (short)0, acc[mt][nt], false, false);
        }
    }
    // ---- + H @ Whh^T ----
    {
        const __bf16* pa = H   + (size_t)(mBase + row) * Fc + kh;
        const __bf16* pb = Whh + (size_t)(nBase + row) * Fc + kh;
#pragma unroll 4
        for (int k = 0; k < Fc; k += 32) {
            v16bf fa[4], fb[2];
#pragma unroll
            for (int mt = 0; mt < 4; ++mt) fa[mt] = load_frag(pa + (size_t)(mt * 16) * Fc + k);
#pragma unroll
            for (int nt = 0; nt < 2; ++nt) fb[nt] = load_frag(pb + (size_t)(nt * 16) * Fc + k);
#pragma unroll
            for (int mt = 0; mt < 4; ++mt)
#pragma unroll
                for (int nt = 0; nt < 2; ++nt)
                    acc[mt][nt] = __builtin_amdgcn_wmma_f32_16x16x32_bf16(
                        false, fa[mt], false, fb[nt], (short)0, acc[mt][nt], false, false);
        }
    }

    // Store: C layout — VGPR r holds M = tile + r + 8*(lane>>4), N = tile + (lane&15)
    const int cn = nBase + (lane & 15);
    const int cm = mBase + ((lane >> 4) << 3);
#pragma unroll
    for (int mt = 0; mt < 4; ++mt) {
#pragma unroll
        for (int nt = 0; nt < 2; ++nt) {
#pragma unroll
            for (int r = 0; r < 8; ++r) {
                gates[(size_t)(cm + mt * 16 + r) * G4F + (cn + nt * 16)] = acc[mt][nt][r];
            }
        }
    }
}

// ---------------- elementwise LSTM update ----------------

__global__ void k_lstm_point(const float* __restrict__ gates,
                             const float* __restrict__ bih,
                             const float* __restrict__ bhh,
                             float* __restrict__ c, float* __restrict__ h,
                             unsigned short* __restrict__ hbf,
                             float* __restrict__ yout, int t) {
    int i = blockIdx.x * blockDim.x + threadIdx.x;
    if (i >= Bc * Fc) return;
    int b = i >> 10;          // / F
    int f = i & (Fc - 1);     // % F
    const float* g = gates + (size_t)b * G4F;

    float ig = g[f]           + bih[f]           + bhh[f];
    float fg = g[Fc + f]      + bih[Fc + f]      + bhh[Fc + f];
    float gg = g[2 * Fc + f]  + bih[2 * Fc + f]  + bhh[2 * Fc + f];
    float og = g[3 * Fc + f]  + bih[3 * Fc + f]  + bhh[3 * Fc + f];

    float si = 1.f / (1.f + __expf(-ig));
    float sf = 1.f / (1.f + __expf(-fg));
    float so = 1.f / (1.f + __expf(-og));
    float tg = tanhf(gg);

    float cn = sf * c[i] + si * tg;
    float hn = so * tanhf(cn);

    c[i] = cn;
    h[i] = hn;
    hbf[i] = f2bf(hn);
    if (yout) yout[(size_t)i * FUTc + t] = hn;   // d_out is [B, F, FUT]
}

// ---------------- host orchestration ----------------

extern "C" void kernel_launch(void* const* d_in, const int* in_sizes, int n_in,
                              void* d_out, int out_size, void* d_ws, size_t ws_size,
                              hipStream_t stream) {
    const float* seq = (const float*)d_in[0];  // [B, F, HIST]
    const float* Wih = (const float*)d_in[1];  // [L, 4F, F]
    const float* Whh = (const float*)d_in[2];  // [L, 4F, F]
    const float* bih = (const float*)d_in[3];  // [L, 4F]
    const float* bhh = (const float*)d_in[4];  // [L, 4F]
    float* out = (float*)d_out;                // [B, F, FUT]

    char* ws = (char*)d_ws;
    size_t off = 0;
    unsigned short* WihB = (unsigned short*)(ws + off); off += (size_t)Lc * G4F * Fc * 2;
    unsigned short* WhhB = (unsigned short*)(ws + off); off += (size_t)Lc * G4F * Fc * 2;
    unsigned short* xB   = (unsigned short*)(ws + off); off += (size_t)Bc * Fc * 2;
    unsigned short* hB   = (unsigned short*)(ws + off); off += (size_t)Lc * Bc * Fc * 2;
    float* hS     = (float*)(ws + off); off += (size_t)Lc * Bc * Fc * 4;
    float* cS     = (float*)(ws + off); off += (size_t)Lc * Bc * Fc * 4;
    float* gatesS = (float*)(ws + off); off += (size_t)Bc * G4F * 4;
    (void)ws_size; (void)in_sizes; (void)n_in; (void)out_size;

    const int nW = Lc * G4F * Fc;                 // 12.58M elems per weight tensor
    const int bfGrid = (Bc * Fc + 255) / 256;     // 2048 blocks
    const int stGrid = (Lc * Bc * Fc + 255) / 256;

    // One-time (per launch) prep: bf16 weights + zero state
    k_f32_to_bf16<<<4096, 256, 0, stream>>>(Wih, WihB, nW);
    k_f32_to_bf16<<<4096, 256, 0, stream>>>(Whh, WhhB, nW);
    k_zero_state<<<stGrid, 256, 0, stream>>>(hS, cS, hB);

    dim3 gemmGrid(G4F / 128, Bc / 128);           // (32, 4) blocks of 256 threads

    // ---- history (encoder) ----
    for (int t = 0; t < HISTc; ++t) {
        k_slice_x<<<bfGrid, 256, 0, stream>>>(seq, xB, t);
        for (int l = 0; l < Lc; ++l) {
            const unsigned short* A = (l == 0) ? xB : hB + (size_t)(l - 1) * Bc * Fc;
            k_gates_gemm<<<gemmGrid, 256, 0, stream>>>(
                A, WihB + (size_t)l * G4F * Fc,
                hB + (size_t)l * Bc * Fc, WhhB + (size_t)l * G4F * Fc, gatesS);
            k_lstm_point<<<bfGrid, 256, 0, stream>>>(
                gatesS, bih + (size_t)l * G4F, bhh + (size_t)l * G4F,
                cS + (size_t)l * Bc * Fc, hS + (size_t)l * Bc * Fc,
                hB + (size_t)l * Bc * Fc, nullptr, 0);
        }
    }

    // ---- decoder: first input is top-layer CELL state after history ----
    k_f32_to_bf16<<<bfGrid, 256, 0, stream>>>(cS + (size_t)2 * Bc * Fc, xB, Bc * Fc);

    for (int t = 0; t < FUTc; ++t) {
        for (int l = 0; l < Lc; ++l) {
            const unsigned short* A =
                (l == 0) ? ((t == 0) ? xB : hB + (size_t)2 * Bc * Fc)
                         : hB + (size_t)(l - 1) * Bc * Fc;
            k_gates_gemm<<<gemmGrid, 256, 0, stream>>>(
                A, WihB + (size_t)l * G4F * Fc,
                hB + (size_t)l * Bc * Fc, WhhB + (size_t)l * G4F * Fc, gatesS);
            k_lstm_point<<<bfGrid, 256, 0, stream>>>(
                gatesS, bih + (size_t)l * G4F, bhh + (size_t)l * G4F,
                cS + (size_t)l * Bc * Fc, hS + (size_t)l * Bc * Fc,
                hB + (size_t)l * Bc * Fc, (l == 2) ? out : nullptr, t);
        }
    }
}
